// MaskedAutoEncoder_51969104281728
// MI455X (gfx1250) — matrix operand order repaired and checked by reading, not compiled, not run on hardware
//
#include <hip/hip_runtime.h>
#include <math.h>

// ---------------------------------------------------------------------------
// MAE decoder forward for MI455X (gfx1250, wave32, WMMA 16x16x32 f16->f32)
// GEMMs: double-buffered LDS tiles, A-panel via GLOBAL_LOAD_ASYNC_TO_LDS_B128
// (ASYNCcnt), fragments via ds_load_b128, 4 back-to-back WMMAs per K-step.
// ---------------------------------------------------------------------------

typedef __attribute__((ext_vector_type(16))) _Float16 v16h;
typedef __attribute__((ext_vector_type(8)))  float    v8f;

#define BATCH    64
#define NPATCH   196
#define NVIS     49
#define ENC_DIM  768
#define DEC_DIM  512
#define FFN_DIM  2048
#define OUT_DIM  3072
#define NHEADS   16
#define HEADD    32
#define NLAYERS  8

union FragU { v16h v; unsigned int u[8]; };

// ---------------------------------------------------------------------------
// f32 -> f16 conversion (weights)
// ---------------------------------------------------------------------------
__global__ void f32_to_f16_kernel(const float* __restrict__ in,
                                  _Float16* __restrict__ out, size_t n) {
  size_t i = (size_t)blockIdx.x * blockDim.x + threadIdx.x;
  size_t stride = (size_t)gridDim.x * blockDim.x;
  for (; i < n; i += stride) out[i] = (_Float16)in[i];
}

// ---------------------------------------------------------------------------
// visible-index + slot map (stable order of unmasked patches)
// ---------------------------------------------------------------------------
__global__ void build_idx_kernel(const unsigned char* __restrict__ mask,
                                 int* __restrict__ vis_idx,
                                 int* __restrict__ slot) {
  int b = threadIdx.x;
  if (b >= BATCH) return;
  int c = 0;
  for (int p = 0; p < NPATCH; ++p) {
    if (mask[b * NPATCH + p] == 0) {
      if (c < NVIS) vis_idx[b * NVIS + c] = p;
      ++c;
    }
    int s = c - 1;
    if (s < 0) s = 0;
    if (s > NVIS - 1) s = NVIS - 1;
    slot[b * NPATCH + p] = s;
  }
}

// ---------------------------------------------------------------------------
// patchify + gather visible -> f16 rows (B*V, 3072), feature order (c, ph, pw)
// ---------------------------------------------------------------------------
__global__ __launch_bounds__(256) void patchify_kernel(
    const float* __restrict__ x, const int* __restrict__ vis_idx,
    _Float16* __restrict__ vis16) {
  const int row = blockIdx.x;            // b*NVIS + v
  const int b = row / NVIS;
  const int p = vis_idx[row];
  const int hp = p / 14, wp = p % 14;
  const int tid = threadIdx.x;
#pragma unroll
  for (int i = 0; i < 12; ++i) {
    int f = tid + i * 256;               // 0..3071
    int c = f >> 8;
    int ph = (f >> 4) & 15;
    int pw = f & 15;
    float v = x[(((size_t)(b * 12 + c)) * 224 + hp * 16 + ph) * 224 + wp * 16 + pw];
    vis16[(size_t)row * OUT_DIM + f] = (_Float16)v;
  }
}

// ---------------------------------------------------------------------------
// Tiled WMMA GEMM: C[M,N] = A[M,K](f16,rowmaj) @ B[K,N](f16,rowmaj) + bias
// Block = 256 threads = 8 waves; C-tile 128x64; wave w -> rows [16w,16w+16).
// Double-buffered LDS; A tile streamed with async global->LDS (ASYNCcnt).
// Epilogue flags: +residual(f32), exact GELU, f32 store, f16 store.
// ---------------------------------------------------------------------------
template <bool ADD_RES, bool GELU, bool W32, bool W16>
__global__ __launch_bounds__(256) void gemm_wmma(
    const _Float16* __restrict__ A, const _Float16* __restrict__ B,
    const float* __restrict__ bias, const float* __restrict__ res,
    float* __restrict__ out32, _Float16* __restrict__ out16,
    int M, int N, int K) {
  __shared__ __align__(16) _Float16 As[2][128 * 32];  // A tile [m][k]
  __shared__ __align__(16) _Float16 Bt[2][64 * 32];   // B tile transposed [n][k]

  const int tid = threadIdx.x;
  const int lane = tid & 31;
  const int wave = tid >> 5;
  const int tileN = blockIdx.x * 64;
  const int tileM = blockIdx.y * 128;

  const int kr = tid >> 3;               // B loader: K row 0..31
  const int nc = (tid & 7) * 8;          // B loader: N chunk 0..56

  v8f acc[4];
#pragma unroll
  for (int j = 0; j < 4; ++j)
#pragma unroll
    for (int r = 0; r < 8; ++r) acc[j][r] = 0.0f;

  // --- A panel: per-thread async global->LDS copies (2 x 16B chunks) ---
  auto issueA = [&](int k0, int buf) {
#pragma unroll
    for (int rr = 0; rr < 2; ++rr) {
      int c = rr * 256 + tid;            // chunk 0..511
      int arow = c >> 2;                 // 0..127
      int kc = (c & 3) * 8;              // 0,8,16,24 halves
      int grow = tileM + arow;
      if (grow > M - 1) grow = M - 1;    // clamp: rows >= M never stored
      unsigned long long ga =
          (unsigned long long)(A + (size_t)grow * K + k0 + kc);
      unsigned lo = (unsigned)(size_t)(&As[buf][arow * 32 + kc]);
      asm volatile("global_load_async_to_lds_b128 %0, %1, off"
                   :: "v"(lo), "v"(ga) : "memory");
    }
  };
  auto loadB = [&](int k0) -> uint4 {
    return *reinterpret_cast<const uint4*>(B + (size_t)(k0 + kr) * N + tileN + nc);
  };
  auto storeB = [&](uint4 v, int buf) {
    union { uint4 q; _Float16 h[8]; } tmp;
    tmp.q = v;
#pragma unroll
    for (int e = 0; e < 8; ++e) Bt[buf][(nc + e) * 32 + kr] = tmp.h[e];
  };

  const int KT = K >> 5;

  // prologue: stage tile 0
  issueA(0, 0);
  storeB(loadB(0), 0);
  asm volatile("s_wait_asynccnt 0x0" ::: "memory");
  __syncthreads();

  for (int kt = 0; kt < KT; ++kt) {
    const int cur = kt & 1;
    const bool hasNext = (kt + 1) < KT;
    uint4 nb;
    if (hasNext) {
      issueA((kt + 1) * 32, 1 - cur);    // async A -> other buffer
      nb = loadB((kt + 1) * 32);         // B in flight during compute
    }

    // ---- fragments (ds_load_b128) ----
    const _Float16* Asb = As[cur];
    const _Float16* Btb = Bt[cur];
    // A (ISA 16-bit A 16x32): lanes 0-15 K{0-7,16-23}, lanes 16-31 K{8-15,24-31}
    const int abase = (wave * 16 + (lane & 15)) * 32 + (lane >> 4) * 8;
    FragU a;
    *reinterpret_cast<uint4*>(&a.u[0]) =
        *reinterpret_cast<const uint4*>(&Asb[abase]);
    *reinterpret_cast<uint4*>(&a.u[4]) =
        *reinterpret_cast<const uint4*>(&Asb[abase + 16]);
    // B: lanes 0-15 K{0-7,16-23}, lanes 16-31 K{8-15,24-31} (matches A)
    const int bbase = (lane & 15) * 32 + (lane >> 4) * 8;
    FragU bf[4];
#pragma unroll
    for (int j = 0; j < 4; ++j) {
      *reinterpret_cast<uint4*>(&bf[j].u[0]) =
          *reinterpret_cast<const uint4*>(&Btb[j * 16 * 32 + bbase]);
      *reinterpret_cast<uint4*>(&bf[j].u[4]) =
          *reinterpret_cast<const uint4*>(&Btb[j * 16 * 32 + bbase + 16]);
    }
#pragma unroll
    for (int j = 0; j < 4; ++j)
      acc[j] = __builtin_amdgcn_wmma_f32_16x16x32_f16(
          false, a.v, false, bf[j].v, (short)0, acc[j], false, false);

    if (hasNext) storeB(nb, 1 - cur);
    asm volatile("s_wait_asynccnt 0x0" ::: "memory");
    __syncthreads();
  }

  // ---- epilogue: D layout: VGPR r -> M = r + (lane>>4)*8, N = lane&15 ----
  const int mloc = (lane >> 4) * 8;
  const int nloc = lane & 15;
#pragma unroll
  for (int j = 0; j < 4; ++j) {
    const int col = tileN + j * 16 + nloc;
    const float bv = bias[col];
#pragma unroll
    for (int r = 0; r < 8; ++r) {
      const int row = tileM + wave * 16 + mloc + r;
      if (row < M) {
        float v = acc[j][r] + bv;
        if (ADD_RES) v += res[(size_t)row * N + col];
        if (GELU) v = 0.5f * v * (1.0f + erff(v * 0.70710678118654752f));
        if (W32) out32[(size_t)row * N + col] = v;
        if (W16) out16[(size_t)row * N + col] = (_Float16)v;
      }
    }
  }
}

// ---------------------------------------------------------------------------
// scatter visible decodings + mask token + pos embed -> h (f32 + f16)
// ---------------------------------------------------------------------------
__global__ __launch_bounds__(256) void scatter_kernel(
    const float* __restrict__ dec_in, const unsigned char* __restrict__ mask,
    const int* __restrict__ slot, const float* __restrict__ mask_token,
    const float* __restrict__ pos_embed, float* __restrict__ h32,
    _Float16* __restrict__ h16) {
  const int row = blockIdx.x;            // b*NPATCH + p
  const int b = row / NPATCH, p = row % NPATCH;
  const int tid = threadIdx.x;
  const bool msk = mask[b * NPATCH + p] != 0;
  const int sl = slot[b * NPATCH + p];
#pragma unroll
  for (int i = 0; i < 2; ++i) {
    int d = tid + i * 256;
    float v = msk ? mask_token[d]
                  : dec_in[((size_t)(b * NVIS + sl)) * DEC_DIM + d];
    v += pos_embed[p * DEC_DIM + d];
    h32[(size_t)row * DEC_DIM + d] = v;
    h16[(size_t)row * DEC_DIM + d] = (_Float16)v;
  }
}

// ---------------------------------------------------------------------------
// attention: one block per (b,h); K,V in LDS; 1 thread/query, online softmax
// ---------------------------------------------------------------------------
__global__ __launch_bounds__(256) void attn_kernel(
    const _Float16* __restrict__ qkv, _Float16* __restrict__ o16) {
  __shared__ float Ks[NPATCH * HEADD];
  __shared__ float Vs[NPATCH * HEADD];
  const int bh = blockIdx.x;
  const int b = bh >> 4;
  const int h = bh & 15;
  const int tid = threadIdx.x;

  for (int i = tid; i < NPATCH * HEADD; i += 256) {
    int s = i >> 5, d = i & 31;
    size_t base = ((size_t)(b * NPATCH + s)) * (3 * DEC_DIM);
    Ks[i] = (float)qkv[base + DEC_DIM + h * HEADD + d];
    Vs[i] = (float)qkv[base + 2 * DEC_DIM + h * HEADD + d];
  }
  __syncthreads();

  if (tid < NPATCH) {
    const int s = tid;
    size_t qb = ((size_t)(b * NPATCH + s)) * (3 * DEC_DIM) + h * HEADD;
    float q[HEADD];
#pragma unroll
    for (int d = 0; d < HEADD; ++d)
      q[d] = (float)qkv[qb + d] * 0.17677669529663687f;   // 1/sqrt(32)
    float m = -3.0e38f, l = 0.0f;
    float o[HEADD];
#pragma unroll
    for (int d = 0; d < HEADD; ++d) o[d] = 0.0f;
    for (int k = 0; k < NPATCH; ++k) {
      float sc = 0.0f;
#pragma unroll
      for (int d = 0; d < HEADD; ++d) sc += q[d] * Ks[k * HEADD + d];
      float mn = fmaxf(m, sc);
      float corr = expf(m - mn);
      float pw = expf(sc - mn);
      l = l * corr + pw;
#pragma unroll
      for (int d = 0; d < HEADD; ++d) o[d] = o[d] * corr + pw * Vs[k * HEADD + d];
      m = mn;
    }
    float inv = 1.0f / l;
    size_t ob = ((size_t)(b * NPATCH + s)) * DEC_DIM + h * HEADD;
#pragma unroll
    for (int d = 0; d < HEADD; ++d) o16[ob + d] = (_Float16)(o[d] * inv);
  }
}

// ---------------------------------------------------------------------------
// layernorm over 512 features; writes f32 + f16
// ---------------------------------------------------------------------------
__global__ __launch_bounds__(256) void layernorm_kernel(
    const float* __restrict__ xin, const float* __restrict__ g,
    const float* __restrict__ bta, float* __restrict__ h32,
    _Float16* __restrict__ h16) {
  __shared__ float red[256];
  const int row = blockIdx.x;
  const int tid = threadIdx.x;
  const size_t base = (size_t)row * DEC_DIM;
  float v0 = xin[base + tid];
  float v1 = xin[base + 256 + tid];
  red[tid] = v0 + v1;
  __syncthreads();
  for (int s = 128; s > 0; s >>= 1) {
    if (tid < s) red[tid] += red[tid + s];
    __syncthreads();
  }
  float mu = red[0] * (1.0f / 512.0f);
  __syncthreads();
  float d0 = v0 - mu, d1 = v1 - mu;
  red[tid] = d0 * d0 + d1 * d1;
  __syncthreads();
  for (int s = 128; s > 0; s >>= 1) {
    if (tid < s) red[tid] += red[tid + s];
    __syncthreads();
  }
  float rstd = rsqrtf(red[0] * (1.0f / 512.0f) + 1e-5f);
  float o0 = d0 * rstd * g[tid] + bta[tid];
  float o1 = d1 * rstd * g[256 + tid] + bta[256 + tid];
  h32[base + tid] = o0;
  h32[base + 256 + tid] = o1;
  h16[base + tid] = (_Float16)o0;
  h16[base + 256 + tid] = (_Float16)o1;
}

// ---------------------------------------------------------------------------
// host-side orchestration
// ---------------------------------------------------------------------------
extern "C" void kernel_launch(void* const* d_in, const int* in_sizes, int n_in,
                              void* d_out, int out_size, void* d_ws,
                              size_t ws_size, hipStream_t stream) {
  (void)in_sizes; (void)n_in; (void)out_size; (void)ws_size;

  const float* x             = (const float*)d_in[0];
  const unsigned char* mask  = (const unsigned char*)d_in[1];
  const float* Wp            = (const float*)d_in[3];
  const float* bp            = (const float*)d_in[4];
  const float* We2d          = (const float*)d_in[5];
  const float* be2d          = (const float*)d_in[6];
  const float* mask_token    = (const float*)d_in[7];
  const float* pos_embed     = (const float*)d_in[8];
  const float* Wqkv          = (const float*)d_in[9];
  const float* bqkv          = (const float*)d_in[10];
  const float* Wo            = (const float*)d_in[11];
  const float* bo            = (const float*)d_in[12];
  const float* ln1_g         = (const float*)d_in[13];
  const float* ln1_b         = (const float*)d_in[14];
  const float* W1            = (const float*)d_in[15];
  const float* b1            = (const float*)d_in[16];
  const float* W2            = (const float*)d_in[17];
  const float* b2            = (const float*)d_in[18];
  const float* ln2_g         = (const float*)d_in[19];
  const float* ln2_b         = (const float*)d_in[20];
  const float* Wr            = (const float*)d_in[21];
  const float* br            = (const float*)d_in[22];

  float* out_rec = (float*)d_out;                                    // (B,196,3072)
  float* out_enc = (float*)d_out + (size_t)BATCH * NPATCH * OUT_DIM; // (B,49,768)

  // ---- workspace arena (≈190 MB, with reuse) ----
  char* ws = (char*)d_ws;
  size_t off = 0;
  auto alloc = [&](size_t bytes) -> char* {
    char* p = ws + off;
    off = (off + bytes + 255) & ~(size_t)255;
    return p;
  };
  _Float16* wp16   = (_Float16*)alloc(2 * (size_t)OUT_DIM * ENC_DIM);
  _Float16* we2d16 = (_Float16*)alloc(2 * (size_t)ENC_DIM * DEC_DIM);
  _Float16* wqkv16 = (_Float16*)alloc(2 * (size_t)NLAYERS * DEC_DIM * 3 * DEC_DIM);
  _Float16* wo16   = (_Float16*)alloc(2 * (size_t)NLAYERS * DEC_DIM * DEC_DIM);
  _Float16* w116   = (_Float16*)alloc(2 * (size_t)NLAYERS * DEC_DIM * FFN_DIM);
  _Float16* w216   = (_Float16*)alloc(2 * (size_t)NLAYERS * FFN_DIM * DEC_DIM);
  _Float16* wr16   = (_Float16*)alloc(2 * (size_t)DEC_DIM * OUT_DIM);
  int* vis_idx     = (int*)alloc(4 * (size_t)BATCH * NVIS);
  int* slot        = (int*)alloc(4 * (size_t)BATCH * NPATCH);
  float* h32       = (float*)alloc(4 * (size_t)BATCH * NPATCH * DEC_DIM);
  float* pre32     = (float*)alloc(4 * (size_t)BATCH * NPATCH * DEC_DIM); // + dec_in
  _Float16* h16    = (_Float16*)alloc(2 * (size_t)BATCH * NPATCH * DEC_DIM);
  _Float16* big1   = (_Float16*)alloc(2 * (size_t)BATCH * NPATCH * FFN_DIM); // vis/qkv/gelu
  _Float16* big2   = (_Float16*)alloc(2 * (size_t)BATCH * NPATCH * DEC_DIM); // enc/attn-out

  _Float16* vis16  = big1;
  _Float16* qkv16  = big1;
  _Float16* gelu16 = big1;
  _Float16* enc16  = big2;
  _Float16* o16    = big2;
  float* dec32     = pre32;

  const int M_vis = BATCH * NVIS;     // 3136
  const int M_all = BATCH * NPATCH;   // 12544

  // ---- weight conversion ----
  f32_to_f16_kernel<<<2048, 256, 0, stream>>>(Wp, wp16, (size_t)OUT_DIM * ENC_DIM);
  f32_to_f16_kernel<<<2048, 256, 0, stream>>>(We2d, we2d16, (size_t)ENC_DIM * DEC_DIM);
  f32_to_f16_kernel<<<2048, 256, 0, stream>>>(Wqkv, wqkv16, (size_t)NLAYERS * DEC_DIM * 3 * DEC_DIM);
  f32_to_f16_kernel<<<2048, 256, 0, stream>>>(Wo, wo16, (size_t)NLAYERS * DEC_DIM * DEC_DIM);
  f32_to_f16_kernel<<<2048, 256, 0, stream>>>(W1, w116, (size_t)NLAYERS * DEC_DIM * FFN_DIM);
  f32_to_f16_kernel<<<2048, 256, 0, stream>>>(W2, w216, (size_t)NLAYERS * FFN_DIM * DEC_DIM);
  f32_to_f16_kernel<<<2048, 256, 0, stream>>>(Wr, wr16, (size_t)DEC_DIM * OUT_DIM);

  // ---- encoder side ----
  build_idx_kernel<<<1, 64, 0, stream>>>(mask, vis_idx, slot);
  patchify_kernel<<<M_vis, 256, 0, stream>>>(x, vis_idx, vis16);

  // encoded = visible @ Wp + bp   (f32 -> output, f16 -> next GEMM)
  gemm_wmma<false, false, true, true>
      <<<dim3(ENC_DIM / 64, (M_vis + 127) / 128), 256, 0, stream>>>(
          vis16, wp16, bp, nullptr, out_enc, enc16, M_vis, ENC_DIM, OUT_DIM);

  // dec_in = encoded @ We2d + be2d
  gemm_wmma<false, false, true, false>
      <<<dim3(DEC_DIM / 64, (M_vis + 127) / 128), 256, 0, stream>>>(
          enc16, we2d16, be2d, nullptr, dec32, nullptr, M_vis, DEC_DIM, ENC_DIM);

  scatter_kernel<<<M_all, 256, 0, stream>>>(dec32, mask, slot, mask_token,
                                            pos_embed, h32, h16);

  // ---- decoder layers ----
  for (int l = 0; l < NLAYERS; ++l) {
    const _Float16* wqkv_l = wqkv16 + (size_t)l * DEC_DIM * 3 * DEC_DIM;
    const _Float16* wo_l   = wo16 + (size_t)l * DEC_DIM * DEC_DIM;
    const _Float16* w1_l   = w116 + (size_t)l * DEC_DIM * FFN_DIM;
    const _Float16* w2_l   = w216 + (size_t)l * FFN_DIM * DEC_DIM;

    // qkv = h @ Wqkv + bqkv (f16)
    gemm_wmma<false, false, false, true>
        <<<dim3(3 * DEC_DIM / 64, M_all / 128), 256, 0, stream>>>(
            h16, wqkv_l, bqkv + (size_t)l * 3 * DEC_DIM, nullptr, nullptr,
            qkv16, M_all, 3 * DEC_DIM, DEC_DIM);

    attn_kernel<<<BATCH * NHEADS, 256, 0, stream>>>(qkv16, o16);

    // pre = o @ Wo + bo + h (residual)
    gemm_wmma<true, false, true, false>
        <<<dim3(DEC_DIM / 64, M_all / 128), 256, 0, stream>>>(
            o16, wo_l, bo + (size_t)l * DEC_DIM, h32, pre32, nullptr, M_all,
            DEC_DIM, DEC_DIM);
    layernorm_kernel<<<M_all, 256, 0, stream>>>(
        pre32, ln1_g + (size_t)l * DEC_DIM, ln1_b + (size_t)l * DEC_DIM, h32, h16);

    // gelu16 = GELU(h @ W1 + b1)
    gemm_wmma<false, true, false, true>
        <<<dim3(FFN_DIM / 64, M_all / 128), 256, 0, stream>>>(
            h16, w1_l, b1 + (size_t)l * FFN_DIM, nullptr, nullptr, gelu16,
            M_all, FFN_DIM, DEC_DIM);

    // pre = gelu @ W2 + b2 + h (residual)
    gemm_wmma<true, false, true, false>
        <<<dim3(DEC_DIM / 64, M_all / 128), 256, 0, stream>>>(
            gelu16, w2_l, b2 + (size_t)l * DEC_DIM, h32, pre32, nullptr, M_all,
            DEC_DIM, FFN_DIM);
    layernorm_kernel<<<M_all, 256, 0, stream>>>(
        pre32, ln2_g + (size_t)l * DEC_DIM, ln2_b + (size_t)l * DEC_DIM, h32, h16);
  }

  // ---- reconstruct = h @ Wr + br ----
  gemm_wmma<false, false, true, false>
      <<<dim3(OUT_DIM / 64, M_all / 128), 256, 0, stream>>>(
          h16, wr16, br, nullptr, out_rec, nullptr, M_all, OUT_DIM, DEC_DIM);
}
